// AlexNetOWT_BN_87299505258585
// MI455X (gfx1250) — 8-bit matrix operand order repaired and checked by reading, compiled, not run on hardware
//
#include <hip/hip_runtime.h>
#include <stdint.h>


typedef __attribute__((ext_vector_type(16))) int   v16i;
typedef __attribute__((ext_vector_type(8)))  float v8f;

union AFrag { v16i v; uint32_t s[16]; uint4 q[4]; };
union CFrag { v8f  v; float    s[8];  };

#define FP8_P1 0x38u
#define FP8_M1 0xB8u

#define BROW 144

__global__ __launch_bounds__(256) void l1_conv_pool_bin(
    const float* __restrict__ x,
    const float* __restrict__ w1,
    const float* __restrict__ b1,
    const float* __restrict__ sw1,
    const float* __restrict__ sb1,
    uint8_t* __restrict__ act1)
{
    int idx = blockIdx.x * 256 + threadIdx.x;
    const int total = 128 * 256 * 482;
    if (idx >= total) return;
    int lp = idx % 482;
    int co = (idx / 482) % 256;
    int b  = idx / (482 * 256);

    const float* xb = x + (size_t)b * 3 * 2560;
    const float* wc = w1 + (size_t)co * 3 * 23;

    float m = -INFINITY;
    for (int j = 0; j < 5; ++j) {
        int l = lp * 5 - 2 + j;
        if (l < 0 || l >= 2406) continue;
        float s = 0.f;
        #pragma unroll
        for (int ci = 0; ci < 3; ++ci) {
            const float* xr = xb + ci * 2560 + l;
            const float* wr = wc + ci * 23;
            #pragma unroll
            for (int k = 0; k < 23; ++k) {
                float ws = (wr[k] >= 0.f) ? 0.1f : -0.1f;
                s += xr[k * 7] * ws;
            }
        }
        s += b1[co];
        m = fmaxf(m, s);
    }
    float ss = (sw1[co] >= 0.f) ? 0.1f : -0.1f;
    float y  = m * ss + sb1[co];
    act1[idx] = (y >= 0.f) ? (uint8_t)FP8_P1 : (uint8_t)FP8_M1;
}

template <int KW>
__global__ __launch_bounds__(256) void pack_weights_fp8(
    const float* __restrict__ w,
    uint32_t* __restrict__ apacked)
{
    const int KCHUNKS = 2 * KW;
    int idx = blockIdx.x * 256 + threadIdx.x;
    if (idx >= KCHUNKS * 16 * 16 * 32) return;
    int v    = idx & 15;
    int lane = (idx >> 4) & 31;
    int mt   = (idx >> 9) & 15;
    int kc   = idx >> 13;

    int m  = mt * 16 + (lane & 15);
    int hh = lane >> 4;
    int vv = v & 7;
    int kbase = (v >> 2) * 32 + (v & 3) * 4 + hh * 16;
    int kw = kc >> 1;
    int ci0 = (kc & 1) * 128;

    uint32_t dw = 0;
    #pragma unroll
    for (int by = 0; by < 4; ++by) {
        int ci = ci0 + kbase + by;
        float wv = w[((size_t)m * 256 + ci) * KW + kw];
        uint32_t code = (wv >= 0.f) ? FP8_P1 : FP8_M1;
        dw |= code << (8 * by);
    }
    apacked[idx] = dw;
}

template <int KW>
__global__ __launch_bounds__(256) void bconv_wmma_fp8(
    const uint32_t* __restrict__ apacked,
    const uint8_t*  __restrict__ act,
    const float*    __restrict__ bias,
    float*          __restrict__ convout,
    int Lin, int Lout, int dil)
{
    const int KCHUNKS = 2 * KW;
    __shared__ __align__(16) uint8_t lbs[32 * BROW];

    const int batch = blockIdx.y;
    const int l0    = blockIdx.x * 32;
    const int t     = threadIdx.x;
    const int wave  = t >> 5;
    const int lane  = t & 31;
    const int h     = lane >> 4;
    const int nn    = lane & 15;

    const uint8_t* actb = act + (size_t)batch * 256 * Lin;

    v8f acc00 = {}, acc01 = {}, acc10 = {}, acc11 = {};

    for (int kc = 0; kc < KCHUNKS; ++kc) {
        const int kw  = kc >> 1;
        const int ci0 = (kc & 1) * 128;

        __syncthreads();
        #pragma unroll
        for (int j = 0; j < 16; ++j) {
            int e  = t + 256 * j;
            int n  = e & 31;
            int ci = e >> 5;
            int li = l0 + n + kw * dil;
            if (li >= Lin) li = Lin - 1;
            lbs[n * BROW + ci] = actb[(size_t)(ci0 + ci) * Lin + li];
        }
        __syncthreads();

        AFrag bf0, bf1;
        const uint8_t* brow0 = lbs + (nn)      * BROW + h * 16;
        const uint8_t* brow1 = lbs + (nn + 16) * BROW + h * 16;
        #pragma unroll
        for (int g = 0; g < 4; ++g) {
            bf0.q[g] = *(const uint4*)(brow0 + g * 32);
            bf1.q[g] = *(const uint4*)(brow1 + g * 32);
        }

        const uint4* a0 = (const uint4*)(apacked +
            (((size_t)kc * 16 + wave)     * 32 + lane) * 16);
        const uint4* a1 = (const uint4*)(apacked +
            (((size_t)kc * 16 + wave + 8) * 32 + lane) * 16);
        AFrag af0, af1;
        #pragma unroll
        for (int g = 0; g < 4; ++g) {
            af0.q[g] = a0[g];
            af1.q[g] = a1[g];
        }

        acc00 = __builtin_amdgcn_wmma_f32_16x16x128_fp8_fp8(
                    af0.v, bf0.v, (short)0, acc00, false, false);
        acc01 = __builtin_amdgcn_wmma_f32_16x16x128_fp8_fp8(
                    af0.v, bf1.v, (short)0, acc01, false, false);
        acc10 = __builtin_amdgcn_wmma_f32_16x16x128_fp8_fp8(
                    af1.v, bf0.v, (short)0, acc10, false, false);
        acc11 = __builtin_amdgcn_wmma_f32_16x16x128_fp8_fp8(
                    af1.v, bf1.v, (short)0, acc11, false, false);
    }

    const int m0 = wave * 16 + h * 8;
    const int m1 = (wave + 8) * 16 + h * 8;
    const int la = l0 + nn;
    const int lb = l0 + 16 + nn;
    CFrag c00, c01, c10, c11;
    c00.v = acc00; c01.v = acc01; c10.v = acc10; c11.v = acc11;
    float* outb = convout + (size_t)batch * 256 * Lout;
    if (la < Lout) {
        #pragma unroll
        for (int r = 0; r < 8; ++r) {
            outb[(size_t)(m0 + r) * Lout + la] = 0.1f * c00.s[r] + bias[m0 + r];
            outb[(size_t)(m1 + r) * Lout + la] = 0.1f * c10.s[r] + bias[m1 + r];
        }
    }
    if (lb < Lout) {
        #pragma unroll
        for (int r = 0; r < 8; ++r) {
            outb[(size_t)(m0 + r) * Lout + lb] = 0.1f * c01.s[r] + bias[m0 + r];
            outb[(size_t)(m1 + r) * Lout + lb] = 0.1f * c11.s[r] + bias[m1 + r];
        }
    }
}

__global__ __launch_bounds__(256) void pool_sb_bact(
    const float* __restrict__ conv,
    const float* __restrict__ sw,
    const float* __restrict__ sb,
    uint8_t* __restrict__ actout,
    int C, int Lc, int Lp, int k, int p)
{
    int idx = blockIdx.x * 256 + threadIdx.x;
    int total = 128 * C * Lp;
    if (idx >= total) return;
    int lp = idx % Lp;
    int c  = (idx / Lp) % C;
    int b  = idx / (Lp * C);

    const float* row = conv + ((size_t)b * C + c) * Lc;
    float m = -INFINITY;
    for (int j = 0; j < k; ++j) {
        int l = lp * k - p + j;
        if (l >= 0 && l < Lc) m = fmaxf(m, row[l]);
    }
    float ss = (sw[c] >= 0.f) ? 0.1f : -0.1f;
    float y  = m * ss + sb[c];
    actout[idx] = (y >= 0.f) ? (uint8_t)FP8_P1 : (uint8_t)FP8_M1;
}

__global__ __launch_bounds__(256) void l6_conv(
    const uint8_t* __restrict__ act5,
    const float*   __restrict__ w6,
    const float*   __restrict__ b6,
    float*         __restrict__ conv6)
{
    int idx = blockIdx.x * 256 + threadIdx.x;
    if (idx >= 128 * 8 * 11) return;
    int l  = idx % 11;
    int co = (idx / 11) % 8;
    int b  = idx / 88;
    int s = 0;
    for (int ci = 0; ci < 256; ++ci) {
        const uint8_t* ar = act5 + ((size_t)b * 256 + ci) * 13 + l;
        const float*   wr = w6 + ((size_t)co * 256 + ci) * 3;
        #pragma unroll
        for (int k = 0; k < 3; ++k) {
            int xs = (ar[k] & 0x80) ? -1 : 1;
            int ws = (wr[k] >= 0.f) ? 1 : -1;
            s += xs * ws;
        }
    }
    conv6[idx] = 0.1f * (float)s + b6[co];
}

__global__ __launch_bounds__(256) void fc1_kernel(
    const uint8_t* __restrict__ act6,
    const float*   __restrict__ fw1,
    const float*   __restrict__ fsw1,
    const float*   __restrict__ fsb1,
    uint8_t* __restrict__ a1)
{
    int idx = blockIdx.x * 256 + threadIdx.x;
    if (idx >= 128 * 512) return;
    int j = idx % 512;
    int b = idx / 512;
    const uint8_t* xr = act6 + (size_t)b * 32;
    const float*   wr = fw1 + (size_t)j * 32;
    int s = 0;
    #pragma unroll
    for (int i = 0; i < 32; ++i) {
        int xs = (xr[i] & 0x80) ? -1 : 1;
        int ws = (wr[i] >= 0.f) ? 1 : -1;
        s += xs * ws;
    }
    float ss = (fsw1[j] >= 0.f) ? 0.1f : -0.1f;
    float y  = (0.1f * (float)s) * ss + fsb1[j];
    a1[idx] = (y >= 0.f) ? (uint8_t)FP8_P1 : (uint8_t)FP8_M1;
}

__global__ __launch_bounds__(256) void fc2_kernel(
    const uint8_t* __restrict__ a1,
    const float*   __restrict__ fw2,
    const float*   __restrict__ fsw2,
    const float*   __restrict__ fsb2,
    float* __restrict__ out)
{
    int idx = blockIdx.x * 256 + threadIdx.x;
    if (idx >= 128 * 1000) return;
    int k = idx % 1000;
    int b = idx / 1000;
    const uint8_t* xr = a1 + (size_t)b * 512;
    const float*   wr = fw2 + (size_t)k * 512;
    int s = 0;
    for (int j = 0; j < 512; ++j) {
        int xs = (xr[j] & 0x80) ? -1 : 1;
        int ws = (wr[j] >= 0.f) ? 1 : -1;
        s += xs * ws;
    }
    float ss = (fsw2[k] >= 0.f) ? 0.1f : -0.1f;
    out[idx] = (0.1f * (float)s) * ss + fsb2[k];
}

static inline int cdiv(int a, int b) { return (a + b - 1) / b; }

extern "C" void kernel_launch(void* const* d_in, const int* in_sizes, int n_in,
                              void* d_out, int out_size, void* d_ws, size_t ws_size,
                              hipStream_t stream)
{
    (void)in_sizes; (void)n_in; (void)out_size; (void)ws_size;
    const float* x   = (const float*)d_in[0];
    const float* w1  = (const float*)d_in[1];
    const float* b1  = (const float*)d_in[2];
    const float* sw1 = (const float*)d_in[3];
    const float* sb1 = (const float*)d_in[4];
    const float* w2  = (const float*)d_in[5];
    const float* b2  = (const float*)d_in[6];
    const float* sw2 = (const float*)d_in[7];
    const float* sb2 = (const float*)d_in[8];
    const float* w3  = (const float*)d_in[9];
    const float* b3  = (const float*)d_in[10];
    const float* sw3 = (const float*)d_in[11];
    const float* sb3 = (const float*)d_in[12];
    const float* w4  = (const float*)d_in[13];
    const float* b4  = (const float*)d_in[14];
    const float* sw4 = (const float*)d_in[15];
    const float* sb4 = (const float*)d_in[16];
    const float* w5  = (const float*)d_in[17];
    const float* b5  = (const float*)d_in[18];
    const float* sw5 = (const float*)d_in[19];
    const float* sb5 = (const float*)d_in[20];
    const float* w6  = (const float*)d_in[21];
    const float* b6  = (const float*)d_in[22];
    const float* sw6 = (const float*)d_in[23];
    const float* sb6 = (const float*)d_in[24];
    const float* fw1  = (const float*)d_in[25];
    const float* fsw1 = (const float*)d_in[26];
    const float* fsb1 = (const float*)d_in[27];
    const float* fw2  = (const float*)d_in[28];
    const float* fsw2 = (const float*)d_in[29];
    const float* fsb2 = (const float*)d_in[30];
    float* out = (float*)d_out;

    uint8_t* wsp = (uint8_t*)d_ws;
    auto carve = [&](size_t bytes) -> uint8_t* {
        uint8_t* r = wsp;
        wsp += (bytes + 255) & ~(size_t)255;
        return r;
    };
    uint8_t*  act1  = carve((size_t)128 * 256 * 482);
    uint32_t* ap2   = (uint32_t*)carve((size_t)26 * 8192 * 4);
    float*    conv2 = (float*)carve((size_t)128 * 256 * 446 * 4);
    uint8_t*  act2  = carve((size_t)128 * 256 * 149);
    uint32_t* ap3   = (uint32_t*)carve((size_t)14 * 8192 * 4);
    float*    conv3 = (float*)carve((size_t)128 * 256 * 137 * 4);
    uint8_t*  act3  = carve((size_t)128 * 256 * 46);
    uint32_t* ap4   = (uint32_t*)carve((size_t)10 * 8192 * 4);
    float*    conv4 = (float*)carve((size_t)128 * 256 * 42 * 4);
    uint8_t*  act4  = carve((size_t)128 * 256 * 42);
    uint32_t* ap5   = (uint32_t*)carve((size_t)10 * 8192 * 4);
    float*    conv5 = (float*)carve((size_t)128 * 256 * 38 * 4);
    uint8_t*  act5  = carve((size_t)128 * 256 * 13);
    float*    conv6 = (float*)carve((size_t)128 * 8 * 11 * 4);
    uint8_t*  act6  = carve((size_t)128 * 8 * 4);
    uint8_t*  a1    = carve((size_t)128 * 512);

    l1_conv_pool_bin<<<cdiv(128 * 256 * 482, 256), 256, 0, stream>>>(
        x, w1, b1, sw1, sb1, act1);

    pack_weights_fp8<13><<<cdiv(26 * 8192, 256), 256, 0, stream>>>(w2, ap2);
    bconv_wmma_fp8<13><<<dim3(cdiv(446, 32), 128), 256, 0, stream>>>(
        ap2, act1, b2, conv2, 482, 446, 3);
    pool_sb_bact<<<cdiv(128 * 256 * 149, 256), 256, 0, stream>>>(
        conv2, sw2, sb2, act2, 256, 446, 149, 3, 1);

    pack_weights_fp8<7><<<cdiv(14 * 8192, 256), 256, 0, stream>>>(w3, ap3);
    bconv_wmma_fp8<7><<<dim3(cdiv(137, 32), 128), 256, 0, stream>>>(
        ap3, act2, b3, conv3, 149, 137, 2);
    pool_sb_bact<<<cdiv(128 * 256 * 46, 256), 256, 0, stream>>>(
        conv3, sw3, sb3, act3, 256, 137, 46, 3, 1);

    pack_weights_fp8<5><<<cdiv(10 * 8192, 256), 256, 0, stream>>>(w4, ap4);
    bconv_wmma_fp8<5><<<dim3(cdiv(42, 32), 128), 256, 0, stream>>>(
        ap4, act3, b4, conv4, 46, 42, 1);
    pool_sb_bact<<<cdiv(128 * 256 * 42, 256), 256, 0, stream>>>(
        conv4, sw4, sb4, act4, 256, 42, 42, 1, 0);

    pack_weights_fp8<5><<<cdiv(10 * 8192, 256), 256, 0, stream>>>(w5, ap5);
    bconv_wmma_fp8<5><<<dim3(cdiv(38, 32), 128), 256, 0, stream>>>(
        ap5, act4, b5, conv5, 42, 38, 1);
    pool_sb_bact<<<cdiv(128 * 256 * 13, 256), 256, 0, stream>>>(
        conv5, sw5, sb5, act5, 256, 38, 13, 3, 1);

    l6_conv<<<cdiv(128 * 8 * 11, 256), 256, 0, stream>>>(act5, w6, b6, conv6);
    pool_sb_bact<<<cdiv(128 * 8 * 4, 256), 256, 0, stream>>>(
        conv6, sw6, sb6, act6, 8, 11, 4, 3, 1);

    fc1_kernel<<<cdiv(128 * 512, 256), 256, 0, stream>>>(act6, fw1, fsw1, fsb1, a1);
    fc2_kernel<<<cdiv(128 * 1000, 256), 256, 0, stream>>>(a1, fw2, fsw2, fsb2, out);
}
